// Transformer_63282048139691
// MI455X (gfx1250) — matrix operand order repaired and checked by reading, not compiled, not run on hardware
//
#include <hip/hip_runtime.h>
#include <hip/hip_bf16.h>

// ---------------------------------------------------------------------------
// Transformer encoder-decoder forward for MI455X (gfx1250).
// All GEMMs in bf16 via V_WMMA_F32_16X16X32_BF16 (fp32 accum); residual/LN/
// softmax in fp32. wave32. LDS double-buffered GEMM; tiles staged by the
// Tensor Data Mover (tensor_load_to_lds + s_wait_tensorcnt) when available,
// register-staged global_load_b128 path otherwise. V is zero-padded to 128
// rows per head so every GEMM runs the unguarded fast path.
// ---------------------------------------------------------------------------

typedef __bf16 bf16;
typedef __attribute__((ext_vector_type(8)))  __bf16 v8bf;
typedef __attribute__((ext_vector_type(16))) __bf16 v16bf;
typedef __attribute__((ext_vector_type(8)))  float  v8f;

#if defined(__has_builtin)
#if __has_builtin(__builtin_amdgcn_tensor_load_to_lds)
#define USE_TDM 1
#endif
#endif
#ifndef USE_TDM
#define USE_TDM 0
#endif

// Problem constants (match reference)
constexpr int Dm  = 1024;   // d_model
constexpr int Hh  = 16;     // heads
constexpr int HDh = 64;     // head dim
constexpr int FFd = 4096;   // ffn dim
constexpr int NEl = 4;
constexpr int NDl = 4;
constexpr int Sl  = 1024;
constexpr int Tl  = 1024;
constexpr int Bb  = 2;
constexpr int Mrows = Sl * Bb;   // 2048
constexpr int Zh  = Bb * Hh;     // 32 batched heads
constexpr int VP  = 128;         // padded V rows (N for PV GEMM)

constexpr int LDS_PITCH = 40;             // 32 data + 8 pad bf16 per row
constexpr int LDS_TILE  = 128 * LDS_PITCH; // elements per buffer

#if USE_TDM
typedef __attribute__((ext_vector_type(4))) unsigned tdm_g0_t;
typedef __attribute__((ext_vector_type(8))) int      tdm_g1_t;
typedef __attribute__((ext_vector_type(4))) int      tdm_g2_t;
#if __clang_major__ >= 23
typedef __attribute__((ext_vector_type(8))) int      tdm_g4_t;
#endif

// Issue one TDM 2D tile load: rows x cols (bf16) from global into LDS with
// 16-byte padding every 64 bytes (matches LDS_PITCH=40 elements).
// D# layout per CDNA5 ISA ch.8 (groups 0/1; groups 2/3 unused for 2D).
__device__ __forceinline__ void tdm_load_2d(unsigned lds_addr, const bf16* gptr,
                                            unsigned rows, unsigned cols,
                                            unsigned tensor_d0, unsigned tensor_d1,
                                            unsigned long stride0 /*elements*/)
{
    unsigned long ga = (unsigned long)(size_t)gptr;
    tdm_g0_t g0;
    g0[0] = 1u;                                   // count=1 (valid descriptor)
    g0[1] = lds_addr;                             // lds_addr
    g0[2] = (unsigned)ga;                         // global_addr[31:0]
    g0[3] = (unsigned)(ga >> 32) | (2u << 30);    // global_addr[56:32] | type=2
    tdm_g1_t g1;
    g1[0] = (1 << 16)        // data_size = 2 bytes
          | (1 << 20)        // pad_enable
          | (3 << 22)        // pad_interval = 16 DWORDs (64B)
          | (3 << 25);       // pad_amount   = 4 DWORDs (16B)
    g1[1] = (int)((tensor_d0 & 0xFFFFu) << 16);                      // dim0 lo
    g1[2] = (int)((tensor_d0 >> 16) | ((tensor_d1 & 0xFFFFu) << 16));// dim0 hi|dim1 lo
    g1[3] = (int)((tensor_d1 >> 16) | (cols << 16));                 // dim1 hi|tile_dim0
    g1[4] = (int)(rows & 0xFFFFu);                                   // tile_dim1
    g1[5] = (int)(unsigned)stride0;                                  // dim0_stride lo
    g1[6] = (int)(unsigned)(stride0 >> 32);                          // dim0_stride hi
    g1[7] = 0;
    tdm_g2_t gz = {0, 0, 0, 0};
#if __clang_major__ >= 23
    tdm_g4_t gz8 = {};
    __builtin_amdgcn_tensor_load_to_lds(g0, g1, gz, gz, gz8, 0);
#else
    __builtin_amdgcn_tensor_load_to_lds(g0, g1, gz, gz, 0);
#endif
}
#endif // USE_TDM

// ---------------------------------------------------------------------------
// GEMM: C[z] = A[z] (MxK row-major bf16) * W[z]^T (W: NxK row-major bf16)
// optional bias[n] (fp32), optional ReLU, fp32 and/or bf16 outputs.
// 256 threads = 8 waves; block tile 128x128; wave tile 64x32; K step 32.
// Requires M%128==0 && N%128==0 && K%32==0 (guaranteed by caller).
// ---------------------------------------------------------------------------
struct TileRegs { v8bf a0, a1, b0, b1; };

__global__ void __launch_bounds__(256)
gemm_bf16_wmma(const bf16* __restrict__ A, long lda, long strideA,
               const bf16* __restrict__ W, long ldw, long strideW,
               const float* __restrict__ bias,
               float* __restrict__ Cf, bf16* __restrict__ Ch,
               long ldc, long strideC,
               int M, int N, int K, int relu)
{
    __shared__ alignas(16) bf16 sA[2][LDS_TILE];
    __shared__ alignas(16) bf16 sB[2][LDS_TILE];

    const int z = blockIdx.z;
    A += (size_t)z * strideA;
    W += (size_t)z * strideW;
    const size_t cOff = (size_t)z * strideC;

    const int n0 = blockIdx.x * 128;
    const int m0 = blockIdx.y * 128;
    const int tid  = threadIdx.x;
    const int lane = tid & 31;
    const int wv   = tid >> 5;
    const int wm   = (wv & 1) * 64;   // wave M offset in tile
    const int wn   = (wv >> 1) * 32;  // wave N offset in tile
    const int half = lane >> 4;       // 0: lanes 0-15, 1: lanes 16-31
    const int l16  = lane & 15;

    v8f acc[4][2] = {};

    auto computeTile = [&](int buf) {
        // A frag: lane<16 -> K0-7 | K16-23 ; lane>=16 -> K8-15 | K24-31
        v16bf af[4];
#pragma unroll
        for (int i = 0; i < 4; ++i) {
            int row = wm + i * 16 + l16;
            const bf16* p = &sA[buf][row * LDS_PITCH + half * 8];
            union { v16bf v; v8bf h[2]; } u;
            u.h[0] = *(const v8bf*)p;
            u.h[1] = *(const v8bf*)(p + 16);
            af[i] = u.v;
        }
        // B frag: lane<16 -> K0-7 | K16-23 ; lane>=16 -> K8-15 | K24-31
        v16bf bfr[2];
#pragma unroll
        for (int j = 0; j < 2; ++j) {
            int row = wn + j * 16 + l16;
            const bf16* p = &sB[buf][row * LDS_PITCH + half * 8];
            union { v16bf v; v8bf h[2]; } u;
            u.h[0] = *(const v8bf*)p;
            u.h[1] = *(const v8bf*)(p + 16);
            bfr[j] = u.v;
        }
#pragma unroll
        for (int i = 0; i < 4; ++i)
#pragma unroll
            for (int j = 0; j < 2; ++j)
                acc[i][j] = __builtin_amdgcn_wmma_f32_16x16x32_bf16(
                    false, af[i], false, bfr[j], (short)0, acc[i][j],
                    false, false);
    };

    const int nk = K >> 5;

#if USE_TDM
    // ---- Tensor Data Mover staging: one wave issues 2 tile DMAs per K-step
    const unsigned ldsA0 = (unsigned)(size_t)&sA[0][0];
    const unsigned ldsB0 = (unsigned)(size_t)&sB[0][0];
    const bf16* aTile = A + (size_t)m0 * lda;
    const bf16* wTile = W + (size_t)n0 * ldw;

    if (wv == 0) {
        tdm_load_2d(ldsA0, aTile, 128, 32, (unsigned)K, (unsigned)M, (unsigned long)lda);
        tdm_load_2d(ldsB0, wTile, 128, 32, (unsigned)K, (unsigned)N, (unsigned long)ldw);
        __builtin_amdgcn_s_wait_tensorcnt(0);
    }
    __syncthreads();
    int cur = 0;
    for (int kt = 0; kt < nk; ++kt) {
        const bool more = (kt + 1) < nk;
        if (more && wv == 0) {
            const int k1 = (kt + 1) << 5;
            const unsigned boff = (unsigned)((cur ^ 1) * LDS_TILE * 2);
            tdm_load_2d(ldsA0 + boff, aTile + k1, 128, 32,
                        (unsigned)K, (unsigned)M, (unsigned long)lda);
            tdm_load_2d(ldsB0 + boff, wTile + k1, 128, 32,
                        (unsigned)K, (unsigned)N, (unsigned long)ldw);
        }
        computeTile(cur);
        if (more) {
            if (wv == 0) __builtin_amdgcn_s_wait_tensorcnt(0);
            __syncthreads();
            cur ^= 1;
        }
    }
#else
    // ---- fallback: register-staged cooperative loads -----------------------
    const int row0 = tid >> 2;
    const int c8   = (tid & 3) * 8;
    const bf16* aptr = A + (size_t)(m0 + row0) * lda + c8;
    const bf16* wptr = W + (size_t)(n0 + row0) * ldw + c8;
    auto loadTile = [&](int k0) -> TileRegs {
        TileRegs t;
        t.a0 = *(const v8bf*)(aptr + k0);
        t.a1 = *(const v8bf*)(aptr + (size_t)64 * lda + k0);
        t.b0 = *(const v8bf*)(wptr + k0);
        t.b1 = *(const v8bf*)(wptr + (size_t)64 * ldw + k0);
        if (k0 + 32 < K) {
            __builtin_prefetch(aptr + k0 + 32, 0, 1);
            __builtin_prefetch(aptr + (size_t)64 * lda + k0 + 32, 0, 1);
            __builtin_prefetch(wptr + k0 + 32, 0, 1);
            __builtin_prefetch(wptr + (size_t)64 * ldw + k0 + 32, 0, 1);
        }
        return t;
    };
    auto storeTile = [&](int buf, const TileRegs& t) {
        *(v8bf*)&sA[buf][(row0)      * LDS_PITCH + c8] = t.a0;
        *(v8bf*)&sA[buf][(row0 + 64) * LDS_PITCH + c8] = t.a1;
        *(v8bf*)&sB[buf][(row0)      * LDS_PITCH + c8] = t.b0;
        *(v8bf*)&sB[buf][(row0 + 64) * LDS_PITCH + c8] = t.b1;
    };
    TileRegs t0 = loadTile(0);
    storeTile(0, t0);
    __syncthreads();
    int cur = 0;
    for (int kt = 0; kt < nk; ++kt) {
        TileRegs nxt;
        const bool more = (kt + 1) < nk;
        if (more) nxt = loadTile((kt + 1) << 5);
        computeTile(cur);
        if (more) {
            storeTile(cur ^ 1, nxt);
            __syncthreads();
            cur ^= 1;
        }
    }
#endif

    // Epilogue. C/D layout: vgpr r, lanes 0-15 -> M=r, lanes 16-31 -> M=8+r.
#pragma unroll
    for (int i = 0; i < 4; ++i) {
        int mBase = m0 + wm + i * 16 + half * 8;
#pragma unroll
        for (int j = 0; j < 2; ++j) {
            int n = n0 + wn + j * 16 + l16;
            float bval = bias ? bias[n] : 0.0f;
#pragma unroll
            for (int r = 0; r < 8; ++r) {
                int m = mBase + r;
                float v = acc[i][j][r] + bval;
                if (relu) v = fmaxf(v, 0.0f);
                size_t off = cOff + (size_t)m * ldc + n;
                if (Cf) Cf[off] = v;
                if (Ch) Ch[off] = (bf16)v;
            }
        }
    }
}

// ---------------------------------------------------------------------------
// Row softmax: 1024-wide rows, one 256-thread block per row. fp32 in, bf16 out
// ---------------------------------------------------------------------------
__global__ void __launch_bounds__(256)
softmax_rows(const float* __restrict__ Sc, bf16* __restrict__ P)
{
    const size_t base = (size_t)blockIdx.x * 1024;
    const int tid = threadIdx.x, lane = tid & 31, wv = tid >> 5;
    float v[4];
    float mx = -3.4e38f;
#pragma unroll
    for (int i = 0; i < 4; ++i) {
        v[i] = Sc[base + tid + i * 256];
        mx = fmaxf(mx, v[i]);
    }
    for (int o = 16; o; o >>= 1) mx = fmaxf(mx, __shfl_xor(mx, o, 32));
    __shared__ float red[8];
    if (!lane) red[wv] = mx;
    __syncthreads();
    if (!tid) {
        float m = red[0];
        for (int i = 1; i < 8; ++i) m = fmaxf(m, red[i]);
        red[0] = m;
    }
    __syncthreads();
    mx = red[0];
    float s = 0.f;
#pragma unroll
    for (int i = 0; i < 4; ++i) { v[i] = __expf(v[i] - mx); s += v[i]; }
    for (int o = 16; o; o >>= 1) s += __shfl_xor(s, o, 32);
    __syncthreads();
    if (!lane) red[wv] = s;
    __syncthreads();
    if (!tid) {
        float t = 0.f;
        for (int i = 0; i < 8; ++i) t += red[i];
        red[0] = t;
    }
    __syncthreads();
    float inv = 1.0f / red[0];
#pragma unroll
    for (int i = 0; i < 4; ++i)
        P[base + tid + i * 256] = (bf16)(v[i] * inv);
}

// ---------------------------------------------------------------------------
// LayerNorm over 1024-wide rows: y = LN(x + resid) * g + b, fp32+bf16 outputs
// ---------------------------------------------------------------------------
__global__ void __launch_bounds__(256)
layernorm_rows(const float* __restrict__ x, const float* __restrict__ resid,
               const float* __restrict__ g, const float* __restrict__ b,
               float* __restrict__ yf, bf16* __restrict__ yh)
{
    const size_t base = (size_t)blockIdx.x * 1024;
    const int tid = threadIdx.x, lane = tid & 31, wv = tid >> 5;
    float v[4];
    float s1 = 0.f, s2 = 0.f;
#pragma unroll
    for (int i = 0; i < 4; ++i) {
        int c = tid + i * 256;
        float t = x[base + c];
        if (resid) t += resid[base + c];
        v[i] = t; s1 += t; s2 += t * t;
    }
    for (int o = 16; o; o >>= 1) {
        s1 += __shfl_xor(s1, o, 32);
        s2 += __shfl_xor(s2, o, 32);
    }
    __shared__ float r1[8], r2[8];
    if (!lane) { r1[wv] = s1; r2[wv] = s2; }
    __syncthreads();
    if (!tid) {
        float a = 0.f, c2 = 0.f;
        for (int i = 0; i < 8; ++i) { a += r1[i]; c2 += r2[i]; }
        r1[0] = a; r2[0] = c2;
    }
    __syncthreads();
    float mean = r1[0] * (1.0f / 1024.0f);
    float var  = r2[0] * (1.0f / 1024.0f) - mean * mean;
    float rstd = rsqrtf(var + 1e-5f);
#pragma unroll
    for (int i = 0; i < 4; ++i) {
        int c = tid + i * 256;
        float o = (v[i] - mean) * rstd * g[c] + b[c];
        if (yf) yf[base + c] = o;
        if (yh) yh[base + c] = (bf16)o;
    }
}

// ---------------------------------------------------------------------------
// fp32 -> bf16 elementwise convert; bf16 zero fill
// ---------------------------------------------------------------------------
__global__ void __launch_bounds__(256)
convert_f32_bf16(const float* __restrict__ in, bf16* __restrict__ out, size_t n)
{
    size_t i = (size_t)blockIdx.x * 256 + threadIdx.x;
    if (i < n) out[i] = (bf16)in[i];
}

__global__ void __launch_bounds__(256)
fill_zero_bf16(bf16* __restrict__ p, size_t n)
{
    size_t i = (size_t)blockIdx.x * 256 + threadIdx.x;
    if (i < n) p[i] = (bf16)0.0f;
}

// ---------------------------------------------------------------------------
// Head packing: [L,B,D] -> per-head layouts for batched attention GEMMs.
// q: [z][L][64] with 1/sqrt(HD) scale; k: [z][L][64]; v: [z][VP][L] transposed
// (rows 64..VP-1 of v stay zero -> unguarded PV GEMM)
// ---------------------------------------------------------------------------
__global__ void __launch_bounds__(256)
pack_q_kernel(const bf16* __restrict__ in, bf16* __restrict__ out, int L, float scale)
{
    size_t idx = (size_t)blockIdx.x * 256 + threadIdx.x;
    size_t total = (size_t)L * Bb * Dm;
    if (idx >= total) return;
    int d = idx % Dm; size_t r = idx / Dm; int b = r % Bb; int t = r / Bb;
    int h = d >> 6, e = d & 63; int z = b * Hh + h;
    out[((size_t)z * L + t) * 64 + e] = (bf16)((float)in[idx] * scale);
}

__global__ void __launch_bounds__(256)
pack_k_kernel(const bf16* __restrict__ in, bf16* __restrict__ out, int L)
{
    size_t idx = (size_t)blockIdx.x * 256 + threadIdx.x;
    size_t total = (size_t)L * Bb * Dm;
    if (idx >= total) return;
    int d = idx % Dm; size_t r = idx / Dm; int b = r % Bb; int t = r / Bb;
    int h = d >> 6, e = d & 63; int z = b * Hh + h;
    out[((size_t)z * L + t) * 64 + e] = in[idx];
}

__global__ void __launch_bounds__(256)
pack_v_kernel(const bf16* __restrict__ in, bf16* __restrict__ out, int L)
{
    size_t idx = (size_t)blockIdx.x * 256 + threadIdx.x;
    size_t total = (size_t)L * Bb * Dm;
    if (idx >= total) return;
    int d = idx % Dm; size_t r = idx / Dm; int b = r % Bb; int t = r / Bb;
    int h = d >> 6, e = d & 63; int z = b * Hh + h;
    out[((size_t)z * VP + e) * L + t] = in[idx];   // transposed: [z][VP][L]
}

__global__ void __launch_bounds__(256)
unpack_o_kernel(const bf16* __restrict__ in, bf16* __restrict__ out, int L)
{
    size_t idx = (size_t)blockIdx.x * 256 + threadIdx.x;
    size_t total = (size_t)L * Bb * Dm;
    if (idx >= total) return;
    int d = idx % Dm; size_t r = idx / Dm; int b = r % Bb; int t = r / Bb;
    int h = d >> 6, e = d & 63; int z = b * Hh + h;
    out[idx] = in[((size_t)z * L + t) * VP + e];
}

// ---------------------------------------------------------------------------
// Host-side orchestration
// ---------------------------------------------------------------------------
extern "C" void kernel_launch(void* const* d_in, const int* in_sizes, int n_in,
                              void* d_out, int out_size, void* d_ws, size_t ws_size,
                              hipStream_t stream)
{
    (void)in_sizes; (void)n_in; (void)out_size; (void)ws_size;

    // ---- inputs (setup_inputs order) ----
    const float* src      = (const float*)d_in[0];
    const float* tgt      = (const float*)d_in[1];
    const float* enc_Wq   = (const float*)d_in[2];
    const float* enc_Wk   = (const float*)d_in[3];
    const float* enc_Wv   = (const float*)d_in[4];
    const float* enc_Wo   = (const float*)d_in[5];
    const float* enc_bo   = (const float*)d_in[6];
    const float* enc_ln1g = (const float*)d_in[7];
    const float* enc_ln1b = (const float*)d_in[8];
    const float* enc_W1   = (const float*)d_in[9];
    const float* enc_b1   = (const float*)d_in[10];
    const float* enc_W2   = (const float*)d_in[11];
    const float* enc_b2   = (const float*)d_in[12];
    const float* enc_ln2g = (const float*)d_in[13];
    const float* enc_ln2b = (const float*)d_in[14];
    const float* enc_fng  = (const float*)d_in[15];
    const float* enc_fnb  = (const float*)d_in[16];
    const float* dec_sWq  = (const float*)d_in[17];
    const float* dec_sWk  = (const float*)d_in[18];
    const float* dec_sWv  = (const float*)d_in[19];
    const float* dec_sWo  = (const float*)d_in[20];
    const float* dec_sbo  = (const float*)d_in[21];
    const float* dec_ln1g = (const float*)d_in[22];
    const float* dec_ln1b = (const float*)d_in[23];
    const float* dec_cWq  = (const float*)d_in[24];
    const float* dec_cWk  = (const float*)d_in[25];
    const float* dec_cWv  = (const float*)d_in[26];
    const float* dec_cWo  = (const float*)d_in[27];
    const float* dec_cbo  = (const float*)d_in[28];
    const float* dec_ln2g = (const float*)d_in[29];
    const float* dec_ln2b = (const float*)d_in[30];
    const float* dec_W1   = (const float*)d_in[31];
    const float* dec_b1   = (const float*)d_in[32];
    const float* dec_W2   = (const float*)d_in[33];
    const float* dec_b2   = (const float*)d_in[34];
    const float* dec_ln3g = (const float*)d_in[35];
    const float* dec_ln3b = (const float*)d_in[36];
    const float* dec_fng  = (const float*)d_in[37];
    const float* dec_fnb  = (const float*)d_in[38];

    // ---- workspace carve-up ----
    size_t cur = 0;
    auto alloc = [&](size_t bytes) -> char* {
        size_t a = (cur + 255) & ~(size_t)255;
        cur = a + bytes;
        return (char*)d_ws + a;
    };
    const size_t DD = (size_t)Dm * Dm;   // 1M
    const size_t FD = (size_t)FFd * Dm;  // 4M

    bf16* hWq_e = (bf16*)alloc(NEl * DD * 2);
    bf16* hWk_e = (bf16*)alloc(NEl * DD * 2);
    bf16* hWv_e = (bf16*)alloc(NEl * DD * 2);
    bf16* hWo_e = (bf16*)alloc(NEl * DD * 2);
    bf16* hW1_e = (bf16*)alloc(NEl * FD * 2);
    bf16* hW2_e = (bf16*)alloc(NEl * FD * 2);
    bf16* hsWq  = (bf16*)alloc(NDl * DD * 2);
    bf16* hsWk  = (bf16*)alloc(NDl * DD * 2);
    bf16* hsWv  = (bf16*)alloc(NDl * DD * 2);
    bf16* hsWo  = (bf16*)alloc(NDl * DD * 2);
    bf16* hcWq  = (bf16*)alloc(NDl * DD * 2);
    bf16* hcWk  = (bf16*)alloc(NDl * DD * 2);
    bf16* hcWv  = (bf16*)alloc(NDl * DD * 2);
    bf16* hcWo  = (bf16*)alloc(NDl * DD * 2);
    bf16* hW1_d = (bf16*)alloc(NDl * FD * 2);
    bf16* hW2_d = (bf16*)alloc(NDl * FD * 2);

    bf16*  xh     = (bf16*)alloc((size_t)Mrows * Dm * 2);
    float* xf     = (float*)alloc((size_t)Mrows * Dm * 4);
    bf16*  qh_t   = (bf16*)alloc((size_t)Mrows * Dm * 2);
    bf16*  kh_t   = (bf16*)alloc((size_t)Mrows * Dm * 2);
    bf16*  vh_t   = (bf16*)alloc((size_t)Mrows * Dm * 2);
    bf16*  qz     = (bf16*)alloc((size_t)Zh * Tl * HDh * 2);
    bf16*  kz     = (bf16*)alloc((size_t)Zh * Sl * HDh * 2);
    bf16*  vzT    = (bf16*)alloc((size_t)Zh * VP * Sl * 2);   // zero-padded
    float* scores = (float*)alloc((size_t)Zh * Tl * Sl * 4);
    bf16*  probs  = (bf16*)alloc((size_t)Zh * Tl * Sl * 2);
    bf16*  oz     = (bf16*)alloc((size_t)Zh * Tl * VP * 2);   // 128-col pitch
    bf16*  att_h  = (bf16*)alloc((size_t)Mrows * Dm * 2);
    float* att_f  = (float*)alloc((size_t)Mrows * Dm * 4);
    bf16*  ffn_h  = (bf16*)alloc((size_t)Mrows * FFd * 2);
    float* ffn_f  = (float*)alloc((size_t)Mrows * Dm * 4);
    bf16*  memh   = (bf16*)alloc((size_t)Mrows * Dm * 2);

    // ---- helpers ----
    auto conv = [&](const float* in, bf16* out, size_t n) {
        convert_f32_bf16<<<dim3((unsigned)((n + 255) / 256)), dim3(256), 0, stream>>>(in, out, n);
    };
    auto gemm = [&](const bf16* A, long lda, long sA,
                    const bf16* W, long ldw, long sW,
                    const float* bias,
                    float* Cf, bf16* Ch, long ldc, long sC,
                    int M, int N, int K, int Z, int relu) {
        dim3 grid((N + 127) / 128, (M + 127) / 128, Z);
        gemm_bf16_wmma<<<grid, dim3(256), 0, stream>>>(
            A, lda, sA, W, ldw, sW, bias, Cf, Ch, ldc, sC, M, N, K, relu);
    };
    auto ln = [&](const float* x, const float* resid, const float* g,
                  const float* b, float* yf, bf16* yh) {
        layernorm_rows<<<dim3(Mrows), dim3(256), 0, stream>>>(x, resid, g, b, yf, yh);
    };
    const unsigned packBlocks = (unsigned)(((size_t)Mrows * Dm + 255) / 256);

    // multi-head attention: query stream xq_h (len Lq), kv stream xkv_h (Lkv)
    auto mha = [&](const bf16* xq_h, int Lq, const bf16* xkv_h, int Lkv,
                   const bf16* Wq, const bf16* Wk, const bf16* Wv,
                   const bf16* Wo, const float* bo, float* out_f) {
        int Mq = Lq * Bb, Mkv = Lkv * Bb;
        gemm(xq_h,  Dm, 0, Wq, Dm, 0, nullptr, nullptr, qh_t, Dm, 0, Mq,  Dm, Dm, 1, 0);
        gemm(xkv_h, Dm, 0, Wk, Dm, 0, nullptr, nullptr, kh_t, Dm, 0, Mkv, Dm, Dm, 1, 0);
        gemm(xkv_h, Dm, 0, Wv, Dm, 0, nullptr, nullptr, vh_t, Dm, 0, Mkv, Dm, Dm, 1, 0);
        pack_q_kernel<<<packBlocks, 256, 0, stream>>>(qh_t, qz, Lq, 0.125f); // 1/sqrt(64)
        pack_k_kernel<<<packBlocks, 256, 0, stream>>>(kh_t, kz, Lkv);
        pack_v_kernel<<<packBlocks, 256, 0, stream>>>(vh_t, vzT, Lkv);
        // scores[z] = q[z] (Lq x 64) * k[z]^T  -> (Lq x Lkv) fp32
        gemm(qz, HDh, (long)Lq * HDh, kz, HDh, (long)Lkv * HDh, nullptr,
             scores, nullptr, Lkv, (long)Lq * Lkv, Lq, Lkv, HDh, Zh, 0);
        softmax_rows<<<dim3(Zh * Lq), dim3(256), 0, stream>>>(scores, probs);
        // o[z] = probs[z] (Lq x Lkv) * v[z]  (v: [VP x Lkv], rows>=64 zero)
        gemm(probs, Lkv, (long)Lq * Lkv, vzT, Lkv, (long)VP * Lkv, nullptr,
             nullptr, oz, VP, (long)Lq * VP, Lq, VP, Lkv, Zh, 0);
        unpack_o_kernel<<<packBlocks, 256, 0, stream>>>(oz, att_h, Lq);
        gemm(att_h, Dm, 0, Wo, Dm, 0, bo, out_f, nullptr, Dm, 0, Mq, Dm, Dm, 1, 0);
    };

    // ---- one-time zero fill for padded V rows (64..127 stay zero) ----
    {
        size_t n = (size_t)Zh * VP * Sl;
        fill_zero_bf16<<<dim3((unsigned)((n + 255) / 256)), dim3(256), 0, stream>>>(vzT, n);
    }

    // ---- convert weights to bf16 (once per call; deterministic) ----
    conv(enc_Wq, hWq_e, NEl * DD);  conv(enc_Wk, hWk_e, NEl * DD);
    conv(enc_Wv, hWv_e, NEl * DD);  conv(enc_Wo, hWo_e, NEl * DD);
    conv(enc_W1, hW1_e, NEl * FD);  conv(enc_W2, hW2_e, NEl * FD);
    conv(dec_sWq, hsWq, NDl * DD);  conv(dec_sWk, hsWk, NDl * DD);
    conv(dec_sWv, hsWv, NDl * DD);  conv(dec_sWo, hsWo, NDl * DD);
    conv(dec_cWq, hcWq, NDl * DD);  conv(dec_cWk, hcWk, NDl * DD);
    conv(dec_cWv, hcWv, NDl * DD);  conv(dec_cWo, hcWo, NDl * DD);
    conv(dec_W1, hW1_d, NDl * FD);  conv(dec_W2, hW2_d, NDl * FD);

    // ================= Encoder =================
    conv(src, xh, (size_t)Mrows * Dm);
    const float* curX = src;
    for (int i = 0; i < NEl; ++i) {
        mha(xh, Sl, xh, Sl,
            hWq_e + i * DD, hWk_e + i * DD, hWv_e + i * DD, hWo_e + i * DD,
            enc_bo + i * Dm, att_f);
        ln(att_f, curX, enc_ln1g + i * Dm, enc_ln1b + i * Dm, xf, xh);
        curX = xf;
        gemm(xh, Dm, 0, hW1_e + i * FD, Dm, 0, enc_b1 + i * FFd,
             nullptr, ffn_h, FFd, 0, Mrows, FFd, Dm, 1, 1 /*relu*/);
        gemm(ffn_h, FFd, 0, hW2_e + i * FD, FFd, 0, enc_b2 + i * Dm,
             ffn_f, nullptr, Dm, 0, Mrows, Dm, FFd, 1, 0);
        ln(ffn_f, xf, enc_ln2g + i * Dm, enc_ln2b + i * Dm, xf, xh);
    }
    // memory = LN(x) -> only bf16 needed (feeds cross-attn GEMMs)
    ln(xf, nullptr, enc_fng, enc_fnb, nullptr, memh);

    // ================= Decoder =================
    conv(tgt, xh, (size_t)Mrows * Dm);
    const float* curY = tgt;
    for (int i = 0; i < NDl; ++i) {
        // self-attention
        mha(xh, Tl, xh, Tl,
            hsWq + i * DD, hsWk + i * DD, hsWv + i * DD, hsWo + i * DD,
            dec_sbo + i * Dm, att_f);
        ln(att_f, curY, dec_ln1g + i * Dm, dec_ln1b + i * Dm, xf, xh);
        curY = xf;
        // cross-attention (kv from encoder memory)
        mha(xh, Tl, memh, Sl,
            hcWq + i * DD, hcWk + i * DD, hcWv + i * DD, hcWo + i * DD,
            dec_cbo + i * Dm, att_f);
        ln(att_f, xf, dec_ln2g + i * Dm, dec_ln2b + i * Dm, xf, xh);
        // FFN
        gemm(xh, Dm, 0, hW1_d + i * FD, Dm, 0, dec_b1 + i * FFd,
             nullptr, ffn_h, FFd, 0, Mrows, FFd, Dm, 1, 1 /*relu*/);
        gemm(ffn_h, FFd, 0, hW2_d + i * FD, FFd, 0, dec_b2 + i * Dm,
             ffn_f, nullptr, Dm, 0, Mrows, Dm, FFd, 1, 0);
        ln(ffn_f, xf, dec_ln3g + i * Dm, dec_ln3b + i * Dm, xf, xh);
    }
    // final norm -> fp32 output
    ln(xf, nullptr, dec_fng, dec_fnb, (float*)d_out, nullptr);
}
